// TemporalAttention_13709535609513
// MI455X (gfx1250) — matrix operand order repaired and checked by reading, not compiled, not run on hardware
//
#include <hip/hip_runtime.h>
#include <hip/hip_bf16.h>

typedef __attribute__((ext_vector_type(16))) _Float16 v16h;
typedef __attribute__((ext_vector_type(8)))  float    v8f;

union Frag {
    v16h v;
    uint4 q[2];
    _Float16 h[16];
};

__device__ __forceinline__ v8f zero8() {
    v8f z = {0.f, 0.f, 0.f, 0.f, 0.f, 0.f, 0.f, 0.f};
    return z;
}

__device__ __forceinline__ v8f wmma_f16(const Frag& a, const Frag& b, v8f c) {
    // v_wmma_f32_16x16x32_f16: D = A(16x32 f16) * B(32x16 f16) + C(16x16 f32)
    return __builtin_amdgcn_wmma_f32_16x16x32_f16(
        /*neg_a=*/false, a.v, /*neg_b=*/false, b.v,
        /*c_mod=*/(short)0, c, /*reuse_a=*/false, /*reuse_b=*/false);
}

// ---------------------------------------------------------------------------
// Kernel 0: convert weights fp32 -> f16 (row-major, L2-resident thereafter)
// ---------------------------------------------------------------------------
__global__ void k_cvt(const float* __restrict__ qkvw, const float* __restrict__ projw,
                      _Float16* __restrict__ w16q, _Float16* __restrict__ w16p) {
    int i = blockIdx.x * 256 + threadIdx.x;
    if (i < 1536 * 512) w16q[i] = (_Float16)qkvw[i];
    if (i < 512 * 512)  w16p[i] = (_Float16)projw[i];
}

// ---------------------------------------------------------------------------
// Kernel 1: GroupNorm. Lanes index contiguous hw (coalesced); each thread
// owns one (B, group): 16ch x 16t = 256-element reduction in registers.
// Writes normalized activations f16, layout XN[n = B*16 + t][c=512].
// ---------------------------------------------------------------------------
__global__ __launch_bounds__(256)
void k_gnorm(const float* __restrict__ x, const float* __restrict__ nw,
             const float* __restrict__ nb, _Float16* __restrict__ XN) {
    const int tid   = threadIdx.x;
    const int hwblk = blockIdx.x;   // 0..3
    const int g     = blockIdx.y;   // 0..31
    const int b     = blockIdx.z;   // 0..1
    const int hw    = hwblk * 256 + tid;
    const size_t base = ((size_t)(b * 512 + g * 16) * 16) * 1024 + hw;

    float s = 0.f, ss = 0.f;
    for (int ci = 0; ci < 16; ++ci)
        for (int ti = 0; ti < 16; ++ti) {
            float v = x[base + (size_t)(ci * 16 + ti) * 1024];
            s += v; ss += v * v;
        }
    float mean = s * (1.f / 256.f);
    float var  = ss * (1.f / 256.f) - mean * mean;
    float inv  = rsqrtf(var + 1e-5f);

    const int B = b * 1024 + hw;
    for (int ti = 0; ti < 16; ++ti) {
        __align__(16) _Float16 hbuf[16];
        for (int ci = 0; ci < 16; ++ci) {
            float v = x[base + (size_t)(ci * 16 + ti) * 1024];
            v = (v - mean) * inv * nw[g * 16 + ci] + nb[g * 16 + ci];
            hbuf[ci] = (_Float16)v;
        }
        uint4* dst = (uint4*)(XN + ((size_t)(B * 16 + ti) * 512 + g * 16));
        const uint4* srcp = (const uint4*)hbuf;
        dst[0] = srcp[0];
        dst[1] = srcp[1];
    }
}

// ---------------------------------------------------------------------------
// Kernel 2: fused QKV-GEMM + attention (rel-pos via WMMA, softmax) + proj-GEMM.
// One workgroup (8 wave32) per B position. Dynamic LDS 72KB (halves):
//   [    0.. 8191] xn_s [16][512]     -- later aliased as a_s [16][512]
//   [ 8192..24575] qk_s [16][16][64]  slot = sidx*8+head (sidx 0=q,1=k)
//       Q-slot of head h reused as: M1 (16x32 f32) then W' (16x32 f16)
//       K-slot of head h reused as: wbuf (16x16 f32) + wp (16x32 f16)
//   [24576..32767] v_s  [8][64][16]
//   [32768..34815] rk16_s [32][64]    rpk rows 49..80 as f16
//   [34816..36863] rv16t_s [64][32]   rpv rows 49..80 transposed, f16
// ---------------------------------------------------------------------------
__global__ __launch_bounds__(256)
void k_fused(const _Float16* __restrict__ XN,
             const _Float16* __restrict__ Wq,
             const _Float16* __restrict__ Wp,
             const float* __restrict__ qkvb,
             const float* __restrict__ projb,
             const float* __restrict__ rpk,
             const float* __restrict__ rpv,
             float* __restrict__ Y) {
    extern __shared__ _Float16 smem[];
    _Float16* xn_s    = smem;                  // 8192 halves
    _Float16* qk_s    = smem + 8192;           // 16384 halves
    _Float16* v_s     = smem + 24576;          // 8192 halves
    _Float16* rk16_s  = smem + 32768;          // 2048 halves [d=32][c=64]
    _Float16* rv16t_s = smem + 34816;          // 2048 halves [ch=64][d=32]

    const int tid  = threadIdx.x;
    const int lane = tid & 31;
    const int wv   = tid >> 5;
    const int Bi   = blockIdx.x;
    const int halfsel = lane >> 4;   // 0 | 1
    const int l16     = lane & 15;

    // ---- stage activation tile + rel-pos windows into LDS ----
    {
        const uint4* src = (const uint4*)(XN + (size_t)Bi * 16 * 512);
        uint4* dst = (uint4*)xn_s;
#pragma unroll
        for (int i = 0; i < 4; ++i) dst[tid + 256 * i] = src[tid + 256 * i];
        for (int i = tid; i < 2048; i += 256) {
            int d = i >> 6, c = i & 63;
            rk16_s[i] = (_Float16)rpk[(size_t)(49 + d) * 64 + c];
            rv16t_s[c * 32 + d] = (_Float16)rpv[(size_t)(49 + d) * 64 + c];
        }
    }
    __syncthreads();

    // ---- GEMM1: qkv[o,t] = Wq(1536x512) @ xn(512x16) + bias; fold scale ----
    {
        v8f acc[12];
#pragma unroll
        for (int j = 0; j < 12; ++j) acc[j] = zero8();

        for (int k = 0; k < 16; ++k) {
            Frag bf;   // B operand: xn[K=c][N=t]
            const uint4* bp = (const uint4*)(xn_s + l16 * 512 + k * 32 + halfsel * 16);
            bf.q[0] = bp[0]; bf.q[1] = bp[1];
#pragma unroll
            for (int j = 0; j < 12; ++j) {
                int o = (wv * 12 + j) * 16 + l16;
                const uint4* ap0 = (const uint4*)(Wq + (size_t)o * 512 + k * 32 + halfsel * 16);
                const uint4* ap1 = (const uint4*)(Wq + (size_t)o * 512 + k * 32 + halfsel * 16 + 8);
                Frag af; af.q[0] = ap0[0]; af.q[1] = ap1[0];
                acc[j] = wmma_f16(af, bf, acc[j]);
            }
        }
        const float scale = 0.35355339059327373f;  // 64^-0.25
#pragma unroll
        for (int j = 0; j < 12; ++j) {
#pragma unroll
            for (int r = 0; r < 8; ++r) {
                int o = (wv * 12 + j) * 16 + r + 8 * halfsel;
                int t = l16;
                float val = acc[j][r] + qkvb[o];
                int sidx = o >> 9;
                int head = (o >> 6) & 7;
                int ch   = o & 63;
                if (sidx < 2) {
                    qk_s[((sidx * 8 + head) * 16 + t) * 64 + ch] = (_Float16)(val * scale);
                } else {
                    v_s[(head * 64 + ch) * 16 + t] = (_Float16)val;
                }
            }
        }
    }
    __syncthreads();

    const int h = wv;                          // wave <-> head
    _Float16* Qh = qk_s + h * 16 * 64;         // [t][64]; later M1 / W' scratch
    _Float16* Kh = qk_s + (8 + h) * 16 * 64;   // [s][64]; later wbuf/wp scratch
    _Float16* Vh = v_s + h * 64 * 16;          // [ch][t]
    float*    wbuf = (float*)Kh;               // 16x16 f32
    _Float16* wp   = Kh + 512;                 // 16x32 f16 (padded softmax W)
    float*    m1b  = (float*)Qh;               // 16x32 f32 (rp-k partial)
    _Float16* wp2  = Qh;                       // 16x32 f16 (W' for rp-v)

    // ---- qk^T (2 WMMA, K=64) and rp-k M1 = Q @ Rk^T (4 WMMA) ----
    v8f wacc = zero8();
    v8f macc[2] = {zero8(), zero8()};
#pragma unroll
    for (int kk = 0; kk < 2; ++kk) {
        const uint4* ap0 = (const uint4*)(Qh + l16 * 64 + kk * 32 + halfsel * 16);
        const uint4* ap1 = (const uint4*)(Qh + l16 * 64 + kk * 32 + halfsel * 16 + 8);
        Frag af; af.q[0] = ap0[0]; af.q[1] = ap1[0];

        const uint4* bp = (const uint4*)(Kh + l16 * 64 + kk * 32 + halfsel * 16);
        Frag bf; bf.q[0] = bp[0]; bf.q[1] = bp[1];
        wacc = wmma_f16(af, bf, wacc);
#pragma unroll
        for (int nt2 = 0; nt2 < 2; ++nt2) {   // N = d window (32)
            const uint4* rp0 = (const uint4*)(rk16_s + (nt2 * 16 + l16) * 64 + kk * 32 + halfsel * 16);
            Frag rf; rf.q[0] = rp0[0]; rf.q[1] = rp0[1];
            macc[nt2] = wmma_f16(af, rf, macc[nt2]);
        }
    }
    __syncthreads();   // Q/K tile reads complete before aliasing as scratch

#pragma unroll
    for (int r = 0; r < 8; ++r) {
        int t = r + 8 * halfsel;
        wbuf[t * 16 + l16] = wacc[r];
        m1b[t * 32 + l16]      = macc[0][r];
        m1b[t * 32 + 16 + l16] = macc[1][r];
    }
    __syncthreads();

    // ---- gather rp-k: w[t,s] += M1[t][s-t+15] ----
#pragma unroll
    for (int i = 0; i < 8; ++i) {
        int e = lane + 32 * i;
        int t = e >> 4, s = e & 15;
        wbuf[e] += m1b[t * 32 + (s - t + 15)];
    }
    __syncthreads();

    // ---- softmax over s (rows handled by lanes 0..15) ----
    if (lane < 16) {
        int t = lane;
        float m = -1e30f;
        for (int s = 0; s < 16; ++s) m = fmaxf(m, wbuf[t * 16 + s]);
        float sum = 0.f;
        for (int s = 0; s < 16; ++s) { float ev = __expf(wbuf[t * 16 + s] - m); wbuf[t * 16 + s] = ev; sum += ev; }
        float inv = 1.f / sum;
        for (int s = 0; s < 16; ++s) wbuf[t * 16 + s] *= inv;
    }
    __syncthreads();

    // ---- pack W (K-pad 16->32) and diagonal-scatter W'[t][d]=w[t,d+t-15] ----
#pragma unroll
    for (int i = 0; i < 8; ++i) {
        int e = lane + 32 * i;
        int t = e >> 4, s = e & 15;
        wp[t * 32 + s]      = (_Float16)wbuf[e];
        wp[t * 32 + 16 + s] = (_Float16)0.f;
    }
#pragma unroll
    for (int i = 0; i < 16; ++i) {
        int e = lane + 32 * i;          // 0..511
        int t = e >> 5, d = e & 31;
        int s = d + t - 15;
        float v = (s >= 0 && s < 16) ? wbuf[t * 16 + s] : 0.f;
        wp2[t * 32 + d] = (_Float16)v;
    }
    __syncthreads();

    // ---- a^T[t,ch] = W@V^T (K=16 pad) + W'@Rv (K=32): 8 WMMAs ----
    {
        _Float16* a_s = xn_s;   // alias: [t][512], col = head*64+ch
        const uint4* ap0 = (const uint4*)(wp + l16 * 32 + halfsel * 16);
        const uint4* ap1 = (const uint4*)(wp + l16 * 32 + halfsel * 16 + 8);
        Frag a1; a1.q[0] = ap0[0]; a1.q[1] = ap1[0];
        const uint4* aq0 = (const uint4*)(wp2 + l16 * 32 + halfsel * 16);
        const uint4* aq1 = (const uint4*)(wp2 + l16 * 32 + halfsel * 16 + 8);
        Frag a2; a2.q[0] = aq0[0]; a2.q[1] = aq1[0];
#pragma unroll
        for (int nt = 0; nt < 4; ++nt) {
            Frag b1;
            if (halfsel == 0) {   // lanes 0..15 carry K=s=0..15; upper K rows zero-pad
                const uint4* bp = (const uint4*)(Vh + (nt * 16 + l16) * 16);
                b1.q[0] = bp[0]; b1.q[1] = bp[1];
            } else {
                b1.q[0] = make_uint4(0, 0, 0, 0);
                b1.q[1] = make_uint4(0, 0, 0, 0);
            }
            const uint4* bq = (const uint4*)(rv16t_s + (nt * 16 + l16) * 32 + halfsel * 16);
            Frag b2; b2.q[0] = bq[0]; b2.q[1] = bq[1];

            v8f acc = wmma_f16(a1, b1, zero8());
            acc = wmma_f16(a2, b2, acc);
#pragma unroll
            for (int r = 0; r < 8; ++r) {
                int t  = r + 8 * halfsel;
                int ch = nt * 16 + l16;
                a_s[t * 512 + h * 64 + ch] = (_Float16)acc[r];
            }
        }
    }
    __syncthreads();

    // ---- proj GEMM (512x512 @ 512x16) + bias -> Y[n][c] (coalesced 32B) ----
    {
        _Float16* a_s = xn_s;
        v8f acc[4];
#pragma unroll
        for (int j = 0; j < 4; ++j) acc[j] = zero8();

        for (int k = 0; k < 16; ++k) {
            const uint4* bp = (const uint4*)(a_s + l16 * 512 + k * 32 + halfsel * 16);
            Frag bf; bf.q[0] = bp[0]; bf.q[1] = bp[1];
#pragma unroll
            for (int j = 0; j < 4; ++j) {
                int o = (wv * 4 + j) * 16 + l16;
                const uint4* ap0 = (const uint4*)(Wp + (size_t)o * 512 + k * 32 + halfsel * 16);
                const uint4* ap1 = (const uint4*)(Wp + (size_t)o * 512 + k * 32 + halfsel * 16 + 8);
                Frag af; af.q[0] = ap0[0]; af.q[1] = ap1[0];
                acc[j] = wmma_f16(af, bf, acc[j]);
            }
        }
#pragma unroll
        for (int j = 0; j < 4; ++j) {
            int cbase = (wv * 4 + j) * 16 + 8 * halfsel;   // 8 consecutive channels
            const float4* pb = (const float4*)(projb + cbase);
            float4 b0 = pb[0], b1 = pb[1];
            float4 lo, hi;
            lo.x = acc[j][0] + b0.x; lo.y = acc[j][1] + b0.y;
            lo.z = acc[j][2] + b0.z; lo.w = acc[j][3] + b0.w;
            hi.x = acc[j][4] + b1.x; hi.y = acc[j][5] + b1.y;
            hi.z = acc[j][6] + b1.z; hi.w = acc[j][7] + b1.w;
            float4* dst = (float4*)(Y + (size_t)(Bi * 16 + l16) * 512 + cbase);
            dst[0] = lo;
            dst[1] = hi;
        }
    }
}

// ---------------------------------------------------------------------------
// Kernel 3: epilogue transpose + residual. Tile: fixed (b,t), 64 c x 128 hw.
// Reads Y[n][c] coalesced (256B rows), LDS transpose (pad-65 conflict-free),
// writes out = x + Y coalesced along hw (512B runs).
// ---------------------------------------------------------------------------
__global__ __launch_bounds__(256)
void k_epi(const float* __restrict__ Y, const float* __restrict__ x,
           float* __restrict__ out) {
    __shared__ float S[128 * 65];
    const int tid   = threadIdx.x;
    const int hwblk = blockIdx.x & 7;
    const int cblk  = blockIdx.x >> 3;
    const int t     = blockIdx.y;
    const int b     = blockIdx.z;
    const int hw0   = hwblk * 128;
    const int c0    = cblk * 64;

#pragma unroll 4
    for (int i = 0; i < 32; ++i) {
        int idx = i * 256 + tid;
        int row = idx >> 6;          // hw offset 0..127
        int cc  = idx & 63;
        S[row * 65 + cc] = Y[((size_t)(b * 1024 + hw0 + row) * 16 + t) * 512 + c0 + cc];
    }
    __syncthreads();
#pragma unroll 4
    for (int j = 0; j < 32; ++j) {
        int idx = j * 256 + tid;
        int cc  = idx >> 7;          // 0..63
        int row = idx & 127;
        size_t o = ((size_t)(b * 512 + c0 + cc) * 16 + t) * 1024 + hw0 + row;
        out[o] = x[o] + S[row * 65 + cc];
    }
}

// ---------------------------------------------------------------------------
extern "C" void kernel_launch(void* const* d_in, const int* in_sizes, int n_in,
                              void* d_out, int out_size, void* d_ws, size_t ws_size,
                              hipStream_t stream) {
    const float* x     = (const float*)d_in[0];
    const float* nw    = (const float*)d_in[1];
    const float* nb    = (const float*)d_in[2];
    const float* qkvw  = (const float*)d_in[3];
    const float* qkvb  = (const float*)d_in[4];
    const float* projw = (const float*)d_in[5];
    const float* projb = (const float*)d_in[6];
    const float* rpk   = (const float*)d_in[7];
    const float* rpv   = (const float*)d_in[8];
    float* out = (float*)d_out;

    _Float16* w16q = (_Float16*)d_ws;            // 1536*512 halves
    _Float16* w16p = w16q + 1536 * 512;          // 512*512 halves
    _Float16* XN   = w16p + 512 * 512;           // 32768*512 halves
    float*    Yb   = (float*)(XN + (size_t)32768 * 512);  // 32768*512 f32

    k_cvt<<<3072, 256, 0, stream>>>(qkvw, projw, w16q, w16p);
    k_gnorm<<<dim3(4, 32, 2), 256, 0, stream>>>(x, nw, nb, XN);
    k_fused<<<2048, 256, 73728, stream>>>(XN, w16q, w16p, qkvb, projb, rpk, rpv, Yb);
    k_epi<<<dim3(64, 16, 2), 256, 0, stream>>>(Yb, x, out);
}
